// CoCRowParallelLinearImpl_44856638439901
// MI455X (gfx1250) — matrix operand order repaired and checked by reading, not compiled, not run on hardware
//
#include <hip/hip_runtime.h>

// ---------------------------------------------------------------------------
// Row-parallel linear as a single GEMM:  out[M,N] = A[M,K] * W[N,K]^T + bias
//   M = SEQ*BATCH = 8192, N = D_MODEL = 4096, K = D_FF = 16384, all fp32.
// Memory-bound (~0.9 GB compulsory traffic @ 23.3 TB/s ~= 40 us >> compute),
// so precision is free: bf16x3 split WMMA (v_wmma_f32_16x16x32_bf16) gives
// ~fp32 accuracy while the matrix pipes stay far from the bottleneck.
// This revision adds a 32x32 supertile block swizzle so the ~1024 co-resident
// workgroups share A/W panels through the 192 MB L2 (traffic ~32 GB -> ~1 GB),
// and forces the global staging loads to stay global_load_b128.
// ---------------------------------------------------------------------------

typedef __attribute__((ext_vector_type(16))) __bf16 v16bf;
typedef __attribute__((ext_vector_type(8)))  __bf16 v8bf;
typedef __attribute__((ext_vector_type(4)))  __bf16 v4bf;
typedef __attribute__((ext_vector_type(8)))  float  v8f;
typedef __attribute__((ext_vector_type(4)))  float  v4f;

#define M_DIM 8192
#define N_DIM 4096
#define K_DIM 16384
#define BM 128
#define BN 128
#define BK 32
#define LDK 40          // BK + 8 bf16 pad -> 80B row stride, kills bank conflicts
#define MT (M_DIM / BM) // 64
#define NT (N_DIM / BN) // 32

union FragU { v16bf v; v8bf h[2]; };
union AccU  { v8f  v; float  f[8]; };

// A-fragment (16x32 bf16, MxK): lane L<16 holds row M=L, K=0..7 in VGPR0-3 and
// K=16..23 in VGPR4-7; lane L>=16 holds row M=L-16, K=8..15 and K=24..31.
__device__ __forceinline__ v16bf frag_a(const __bf16* t, int row0, int lane) {
    const int r  = row0 + (lane & 15);
    const int ko = (lane >> 4) * 8;
    FragU u;
    u.h[0] = *(const v8bf*)(t + r * LDK + ko);       // 16B-aligned ds_load_b128
    u.h[1] = *(const v8bf*)(t + r * LDK + ko + 16);
    return u.v;
}

// B-fragment (32x16 bf16, KxN): lane = column N, K along VGPRs;
// lanes 0-15 hold K=0..7,16..23, lanes 16-31 hold K=8..15,24..31. Source sB is [N][K].
__device__ __forceinline__ v16bf frag_b(const __bf16* t, int col0, int lane) {
    const int c  = col0 + (lane & 15);
    const int kb = (lane >> 4) * 8;
    FragU u;
    u.h[0] = *(const v8bf*)(t + c * LDK + kb);
    u.h[1] = *(const v8bf*)(t + c * LDK + kb + 16);
    return u.v;
}

__global__ __launch_bounds__(256)
void rowpar_linear_bf16x3(const float* __restrict__ A,
                          const float* __restrict__ W,
                          const float* __restrict__ bias,
                          float* __restrict__ out) {
    __shared__ __bf16 sAhi[BM * LDK];
    __shared__ __bf16 sAlo[BM * LDK];
    __shared__ __bf16 sBhi[BN * LDK];
    __shared__ __bf16 sBlo[BN * LDK];

    const int tid  = threadIdx.x;
    const int lane = tid & 31;
    const int wave = tid >> 5;          // 8 wave32s
    const int wm   = wave >> 1;         // 0..3 : 32-row slab
    const int wn   = wave & 1;          // 0..1 : 64-col slab

    // ---- 32x32 supertile swizzle over the 1D grid (2048 blocks) ----
    // Each group of 1024 consecutive blocks (the co-resident set) covers a
    // 32(M) x 32(N) tile square; tm varies fastest so consecutive blocks
    // share the same W panel, and all 32 A panels of the group stay L2-hot.
    const int pid    = blockIdx.x;
    const int super  = pid >> 10;          // 0..1
    const int within = pid & 1023;
    const int tm     = super * 32 + (within & 31);
    const int tn     = within >> 5;
    const int m0     = tm * BM;
    const int n0     = tn * BN;

    const float* aBase = A + (size_t)m0 * K_DIM;
    const float* bBase = W + (size_t)n0 * K_DIM;

    // Staging plan: 128 rows x 8 float4 per tile = 1024 float4 / 256 threads = 4 each.
    int rowS[4], colS[4];
#pragma unroll
    for (int i = 0; i < 4; ++i) {
        const int g = tid + i * 256;
        rowS[i] = g >> 3;
        colS[i] = (g & 7) * 4;
    }

    v8f acc[2][4] = {};

    for (int kc = 0; kc < K_DIM / BK; ++kc) {
        const int kofs = kc * BK;

        // ---- global stage: f32 tiles via b128 vector loads ----
        v4f ar[4], br[4];
#pragma unroll
        for (int i = 0; i < 4; ++i) {
            ar[i] = *(const v4f*)(aBase + (size_t)rowS[i] * K_DIM + kofs + colS[i]);
            br[i] = *(const v4f*)(bBase + (size_t)rowS[i] * K_DIM + kofs + colS[i]);
        }
        if (kc + 1 < K_DIM / BK) {       // uniform branch; prefetch next K-chunk
#pragma unroll
            for (int i = 0; i < 4; ++i) {
                __builtin_prefetch(aBase + (size_t)rowS[i] * K_DIM + kofs + BK + colS[i], 0, 0);
                __builtin_prefetch(bBase + (size_t)rowS[i] * K_DIM + kofs + BK + colS[i], 0, 0);
            }
        }

        __syncthreads();                 // previous iteration's LDS reads done

        // ---- split f32 -> bf16 hi + bf16 lo and stage to LDS (b64 stores) ----
#pragma unroll
        for (int i = 0; i < 4; ++i) {
            v4bf ah, al, bh, bl;
#pragma unroll
            for (int e = 0; e < 4; ++e) {
                const float ax = ar[i][e];
                const float bx = br[i][e];
                const __bf16 h = (__bf16)ax;
                ah[e] = h;
                al[e] = (__bf16)(ax - (float)h);
                const __bf16 g = (__bf16)bx;
                bh[e] = g;
                bl[e] = (__bf16)(bx - (float)g);
            }
            const int o = rowS[i] * LDK + colS[i];
            *(v4bf*)(sAhi + o) = ah;
            *(v4bf*)(sAlo + o) = al;
            *(v4bf*)(sBhi + o) = bh;
            *(v4bf*)(sBlo + o) = bl;
        }

        __syncthreads();                 // LDS writes visible

        // ---- WMMA compute: 2x4 tiles of 16x16, bf16x3 correction ----
        v16bf fa_h[2], fa_l[2], fb_h[4], fb_l[4];
#pragma unroll
        for (int i = 0; i < 2; ++i) {
            fa_h[i] = frag_a(sAhi, wm * 32 + i * 16, lane);
            fa_l[i] = frag_a(sAlo, wm * 32 + i * 16, lane);
        }
#pragma unroll
        for (int j = 0; j < 4; ++j) {
            fb_h[j] = frag_b(sBhi, wn * 64 + j * 16, lane);
            fb_l[j] = frag_b(sBlo, wn * 64 + j * 16, lane);
        }
#pragma unroll
        for (int i = 0; i < 2; ++i) {
#pragma unroll
            for (int j = 0; j < 4; ++j) {
                acc[i][j] = __builtin_amdgcn_wmma_f32_16x16x32_bf16(
                    false, fa_h[i], false, fb_h[j], (short)0, acc[i][j], false, false);
                acc[i][j] = __builtin_amdgcn_wmma_f32_16x16x32_bf16(
                    false, fa_h[i], false, fb_l[j], (short)0, acc[i][j], false, false);
                acc[i][j] = __builtin_amdgcn_wmma_f32_16x16x32_bf16(
                    false, fa_l[i], false, fb_h[j], (short)0, acc[i][j], false, false);
            }
        }
    }

    // ---- epilogue: C/D layout -> VGPR r holds row r (lanes 0-15) / r+8 (16-31) ----
#pragma unroll
    for (int i = 0; i < 2; ++i) {
#pragma unroll
        for (int j = 0; j < 4; ++j) {
            AccU u; u.v = acc[i][j];
            const int col = n0 + wn * 64 + j * 16 + (lane & 15);
            const int row = m0 + wm * 32 + i * 16 + ((lane >> 4) * 8);
            const float b = bias[col];
#pragma unroll
            for (int r = 0; r < 8; ++r) {
                out[(size_t)(row + r) * N_DIM + col] = u.f[r] + b;
            }
        }
    }
}

extern "C" void kernel_launch(void* const* d_in, const int* in_sizes, int n_in,
                              void* d_out, int out_size, void* d_ws, size_t ws_size,
                              hipStream_t stream) {
    (void)in_sizes; (void)n_in; (void)out_size; (void)d_ws; (void)ws_size;
    const float* A    = (const float*)d_in[0];   // [8192, 16384] (S*B, D_FF)
    const float* W    = (const float*)d_in[1];   // [4096, 16384] (D_MODEL, D_FF)
    const float* bias = (const float*)d_in[2];   // [4096]
    float*       out  = (float*)d_out;           // [8192, 4096]

    dim3 grid(MT * NT);                          // 2048 blocks, swizzled in-kernel
    dim3 block(256);
    rowpar_linear_bf16x3<<<grid, block, 0, stream>>>(A, W, bias, out);
}
